// ApproximateAttention_39530878992410
// MI455X (gfx1250) — matrix operand order repaired and checked by reading, not compiled, not run on hardware
//
#include <hip/hip_runtime.h>
#include <math.h>

#define BS     16
#define SEQ    4096
#define EMB    64
#define NH     8
#define NBUCK  64
#define HPB    512            // NH * NBUCK
#define HSEQ   (NH * SEQ)     // 32768 sorted entries per batch
#define BINSZ  128            // tokens per bin
#define RR     2.5f
#define TPB    16             // tokens per hash block

typedef __attribute__((ext_vector_type(16))) _Float16 v16h;
typedef __attribute__((ext_vector_type(8)))  _Float16 v8h;
typedef __attribute__((ext_vector_type(4)))  _Float16 v4h;
typedef __attribute__((ext_vector_type(8)))  float    v8f;

// ---------------------------------------------------------------------------
// WMMA fragment loaders (CDNA5 16-bit layouts, CDNA5 ISA §7.12.2)
// A 16x32 f16: lane m=l&15, hi=l>>4; elems 0..7 -> K = kbase+8*hi+e,
//                                    elems 8..15 -> K = kbase+16+8*hi+(e-8)
// ---------------------------------------------------------------------------
static __device__ __forceinline__ v16h frag_A(const _Float16* base, int ld,
                                              int row_base, int kbase, int lane) {
  int m = lane & 15, hi = lane >> 4;
  const _Float16* p = base + (row_base + m) * ld + kbase + hi * 8;
  v8h lo = *(const v8h*)p;
  v8h up = *(const v8h*)(p + 16);
  v16h r;
#pragma unroll
  for (int i = 0; i < 8; ++i) { r[i] = lo[i]; r[i + 8] = up[i]; }
  return r;
}

// B 32x16 f16: lane n=l&15, hi=l>>4; elems 0..7 -> K = kbase+8*hi+e, elems 8..15 -> K = kbase+16+8*hi+(e-8).
// Source laid out so K is contiguous: src[(nbase+n)*ld + kbase + k].
static __device__ __forceinline__ v16h frag_B(const _Float16* base, int ld,
                                              int nbase, int kbase, int lane) {
  int n = lane & 15, hi = lane >> 4;
  const _Float16* p = base + (nbase + n) * ld + kbase + hi * 8;
  v8h lo = *(const v8h*)p;
  v8h up = *(const v8h*)(p + 16);
  v16h r;
#pragma unroll
  for (int i = 0; i < 8; ++i) { r[i] = lo[i]; r[i + 8] = up[i]; }
  return r;
}

// ---------------------------------------------------------------------------
// K0: zero the output (harness poisons d_out; we accumulate into it)
// ---------------------------------------------------------------------------
__global__ void k_zero(float* __restrict__ p, int n) {
  int i = blockIdx.x * blockDim.x + threadIdx.x;
  if (i < n) p[i] = 0.0f;
}

// ---------------------------------------------------------------------------
// K1: ALSH hash, 16 tokens per block (512 thr) to amortize alpha traffic.
// Phase 1: one wave per token builds the 67-dim extended vector (f32).
// Phase 2: thread t owns projection column t, 16 register accumulators;
//          each alpha element is loaded once and reused 16x.
// Phase 3: 128 (token,hash) pairs reduce 64 floored values with weights.
// ---------------------------------------------------------------------------
__global__ void __launch_bounds__(512) k_hash(const float* __restrict__ x,
                                              const float* __restrict__ alpha,
                                              const float* __restrict__ beta,
                                              const float* __restrict__ bw,
                                              unsigned char* __restrict__ bkt,
                                              int is_key) {
  __shared__ float vecs[TPB][68];
  __shared__ float hsh[TPB][HPB];          // 32 KB
  __shared__ float bws[NBUCK];
  int t = threadIdx.x;
  int j = t >> 5, lane = t & 31;           // wave j handles token j
  int tok0 = blockIdx.x * TPB;
  {
    int tok = tok0 + j;
    const float* xp = x + (size_t)tok * EMB;
    float x0 = xp[lane], x1 = xp[lane + 32];
    float ss = x0 * x0 + x1 * x1;
#pragma unroll
    for (int o = 1; o < 32; o <<= 1) ss += __shfl_xor(ss, o, 32);
    if (is_key) {
      vecs[j][lane] = x0; vecs[j][lane + 32] = x1;
      if (lane == 0) {
        vecs[j][64] = ss;                       // norm^2
        vecs[j][65] = ss * ss;                  // norm^4
        vecs[j][66] = ss * ss * ss * ss;        // norm^8
      }
    } else {
      float nrm = sqrtf(ss);
      vecs[j][lane] = x0 / nrm; vecs[j][lane + 32] = x1 / nrm;
      if (lane == 0) { vecs[j][64] = 0.5f; vecs[j][65] = 0.5f; vecs[j][66] = 0.5f; }
    }
  }
  if (t < NBUCK) bws[t] = bw[t];
  __syncthreads();
  {
    float acc[TPB];
#pragma unroll
    for (int j2 = 0; j2 < TPB; ++j2) acc[j2] = 0.0f;
    for (int i = 0; i < 67; ++i) {
      float a = alpha[i * HPB + t];            // loaded once, reused 16x
#pragma unroll
      for (int j2 = 0; j2 < TPB; ++j2) acc[j2] += vecs[j2][i] * a;
    }
    float be = beta[t];
#pragma unroll
    for (int j2 = 0; j2 < TPB; ++j2) hsh[j2][t] = floorf((acc[j2] + be) / RR);
  }
  __syncthreads();
  if (t < TPB * NH) {
    int j2 = t >> 3, hh = t & 7;
    float s = 0.0f;
    for (int k = 0; k < NBUCK; ++k) s += hsh[j2][hh * NBUCK + k] * bws[k];
    int bb = (int)s;                           // astype(int32): trunc toward 0
    int r = bb % NBUCK; if (r < 0) r += NBUCK; // jnp mod semantics
    int tok = tok0 + j2;
    int b = tok / SEQ, n = tok % SEQ;
    bkt[((size_t)b * NH + hh) * SEQ + n] = (unsigned char)r;
  }
}

// ---------------------------------------------------------------------------
// K2: stable counting sort of one (batch, hash) segment of 4096 entries by
// bucket (64 buckets). Emits sorted source token ids and (optionally) the
// undo-sort index (position within the batch's 32768 sorted entries).
// ---------------------------------------------------------------------------
__global__ void __launch_bounds__(256) k_sort(const unsigned char* __restrict__ bkt,
                                              int* __restrict__ srctok,
                                              int* __restrict__ undo) {
  __shared__ unsigned char bv[SEQ];
  __shared__ int cnt[NBUCK];
  __shared__ int off[NBUCK];
  int t = threadIdx.x;
  int seg = blockIdx.x;                      // b*NH + h
  if (t < NBUCK) cnt[t] = 0;
  __syncthreads();
  const unsigned char* bp = bkt + (size_t)seg * SEQ;
  for (int i = t; i < SEQ; i += 256) {
    unsigned char v = bp[i];
    bv[i] = v;
    atomicAdd(&cnt[v], 1);
  }
  __syncthreads();
  if (t == 0) {
    int acc = 0;
    for (int i = 0; i < NBUCK; ++i) { off[i] = acc; acc += cnt[i]; }
  }
  __syncthreads();
  if (t < NBUCK) {
    int pos = off[t];
    int h = seg % NH;
    int base = seg * SEQ;                    // (b*NH+h)*SEQ
    int segoff = h * SEQ;                    // within-batch sorted offset
    for (int n = 0; n < SEQ; ++n) {
      if (bv[n] == (unsigned char)t) {
        srctok[base + pos] = n;
        if (undo) undo[base + n] = segoff + pos;
        ++pos;
      }
    }
  }
}

// ---------------------------------------------------------------------------
// K3: per-bin row logsumexp of S = Q K^T (128x128, E=64) via WMMA f16.
// One block (8 waves) per (batch, bin). Each wave owns 16 rows.
// Gathers use float4 global loads + 64-bit DS stores.
// ---------------------------------------------------------------------------
__global__ void __launch_bounds__(256) k_lse(const float* __restrict__ query,
                                             const float* __restrict__ key,
                                             const int* __restrict__ srcq,
                                             const int* __restrict__ srck,
                                             float* __restrict__ lse) {
  __shared__ __align__(16) _Float16 sQ[BINSZ * EMB];
  __shared__ __align__(16) _Float16 sK[BINSZ * EMB];
  __shared__ int tq[BINSZ], tk[BINSZ];
  int t = threadIdx.x;
  int b = blockIdx.x >> 8, bin = blockIdx.x & 255;
  int sbase = bin * BINSZ;                   // sorted position within batch
  if (t < BINSZ) {
    tq[t] = srcq[(size_t)b * HSEQ + sbase + t];
    tk[t] = srck[(size_t)b * HSEQ + sbase + t];
  }
  __syncthreads();
  for (int idx = t; idx < BINSZ * (EMB / 4); idx += 256) {
    int i = idx >> 4, c4 = (idx & 15) * 4;   // token i, dim chunk c4..c4+3
    float4 qv = *(const float4*)&query[((size_t)b * SEQ + tq[i]) * EMB + c4];
    float4 kv = *(const float4*)&key [((size_t)b * SEQ + tk[i]) * EMB + c4];
    v4h hq = {(_Float16)qv.x, (_Float16)qv.y, (_Float16)qv.z, (_Float16)qv.w};
    v4h hk = {(_Float16)kv.x, (_Float16)kv.y, (_Float16)kv.z, (_Float16)kv.w};
    *(v4h*)&sQ[i * EMB + c4] = hq;
    *(v4h*)&sK[i * EMB + c4] = hk;
  }
  __syncthreads();
  int wv = t >> 5, lane = t & 31;
  int rb = wv * 16;
  v8f acc[8];
#pragma unroll
  for (int c = 0; c < 8; ++c) { v8f z = {}; acc[c] = z; }
#pragma unroll
  for (int ks = 0; ks < EMB; ks += 32) {
    v16h a = frag_A(sQ, EMB, rb, ks, lane);
#pragma unroll
    for (int c = 0; c < 8; ++c) {
      v16h bmat = frag_B(sK, EMB, c * 16, ks, lane);
      acc[c] = __builtin_amdgcn_wmma_f32_16x16x32_f16(
          false, a, false, bmat, (short)0, acc[c], false, false);
    }
  }
  // rowwise logsumexp: row m = rb + v + 8*(lane>>4); cols spread across
  // 16 lanes (n = lane&15) x 8 tiles.
#pragma unroll
  for (int v = 0; v < 8; ++v) {
    float mx = acc[0][v];
#pragma unroll
    for (int c = 1; c < 8; ++c) mx = fmaxf(mx, acc[c][v]);
#pragma unroll
    for (int o = 1; o < 16; o <<= 1) mx = fmaxf(mx, __shfl_xor(mx, o, 32));
    float sm = 0.0f;
#pragma unroll
    for (int c = 0; c < 8; ++c) sm += __expf(acc[c][v] - mx);
#pragma unroll
    for (int o = 1; o < 16; o <<= 1) sm += __shfl_xor(sm, o, 32);
    if ((lane & 15) == 0) {
      int row = rb + v + 8 * (lane >> 4);
      lse[(size_t)b * HSEQ + sbase + row] = mx + __logf(sm);
    }
  }
}

// ---------------------------------------------------------------------------
// K4: per-token logsumexp over the 8 hash rounds: LSEtot[b][n].
// ---------------------------------------------------------------------------
__global__ void k_lsetot(const int* __restrict__ undoq,
                         const float* __restrict__ lse,
                         float* __restrict__ lt) {
  int idx = blockIdx.x * blockDim.x + threadIdx.x;   // b*SEQ + n
  if (idx >= BS * SEQ) return;
  int b = idx / SEQ, n = idx % SEQ;
  float l[NH];
  float mx = -1e30f;
#pragma unroll
  for (int h = 0; h < NH; ++h) {
    int p = undoq[((size_t)b * NH + h) * SEQ + n];
    l[h] = lse[(size_t)b * HSEQ + p];
    mx = fmaxf(mx, l[h]);
  }
  float s = 0.0f;
#pragma unroll
  for (int h = 0; h < NH; ++h) s += __expf(l[h] - mx);
  lt[idx] = mx + __logf(s);
}

// ---------------------------------------------------------------------------
// K5: full attention pass for ONE hash round h (launched 8x, serialized on
// the stream so out-accumulation is race-free & deterministic):
//   out[b,tok,:] += exp(S - LSEtot[tok]) @ V        (folds softmax + merge w)
// One block per (batch, bin-within-hash). P overlays Q/K LDS after barrier.
// ---------------------------------------------------------------------------
__global__ void __launch_bounds__(256) k_attn(const float* __restrict__ query,
                                              const float* __restrict__ key,
                                              const float* __restrict__ value,
                                              const int* __restrict__ srcq,
                                              const int* __restrict__ srck,
                                              const float* __restrict__ lt,
                                              float* __restrict__ out, int h) {
  __shared__ __align__(16) _Float16 sQK[2 * BINSZ * EMB]; // Q|K, later P(128x128)
  __shared__ __align__(16) _Float16 sVt[EMB * BINSZ];     // V transposed [d][i]
  __shared__ int tq[BINSZ], tk[BINSZ];
  int t = threadIdx.x;
  int b = blockIdx.x >> 5, binl = blockIdx.x & 31;
  int sbase = h * SEQ + binl * BINSZ;
  _Float16* sQ = sQK;
  _Float16* sK = sQK + BINSZ * EMB;
  if (t < BINSZ) {
    tq[t] = srcq[(size_t)b * HSEQ + sbase + t];
    tk[t] = srck[(size_t)b * HSEQ + sbase + t];
  }
  __syncthreads();
  for (int idx = t; idx < BINSZ * (EMB / 4); idx += 256) {
    int i = idx >> 4, c4 = (idx & 15) * 4;   // token i, dim chunk c4..c4+3
    float4 qv = *(const float4*)&query[((size_t)b * SEQ + tq[i]) * EMB + c4];
    float4 kv = *(const float4*)&key [((size_t)b * SEQ + tk[i]) * EMB + c4];
    float4 vv = *(const float4*)&value[((size_t)b * SEQ + tk[i]) * EMB + c4];
    v4h hq = {(_Float16)qv.x, (_Float16)qv.y, (_Float16)qv.z, (_Float16)qv.w};
    v4h hk = {(_Float16)kv.x, (_Float16)kv.y, (_Float16)kv.z, (_Float16)kv.w};
    *(v4h*)&sQ[i * EMB + c4] = hq;
    *(v4h*)&sK[i * EMB + c4] = hk;
    sVt[(c4 + 0) * BINSZ + i] = (_Float16)vv.x;   // transposed scatter
    sVt[(c4 + 1) * BINSZ + i] = (_Float16)vv.y;
    sVt[(c4 + 2) * BINSZ + i] = (_Float16)vv.z;
    sVt[(c4 + 3) * BINSZ + i] = (_Float16)vv.w;
  }
  int wv = t >> 5, lane = t & 31;
  int rb = wv * 16, hi = lane >> 4, nn = lane & 15;
  float ltr[8];
#pragma unroll
  for (int v = 0; v < 8; ++v)
    ltr[v] = lt[(size_t)b * SEQ + tq[rb + v + 8 * hi]];
  __syncthreads();

  // S = Q K^T for this wave's 16 rows
  v8f acc[8];
#pragma unroll
  for (int c = 0; c < 8; ++c) { v8f z = {}; acc[c] = z; }
#pragma unroll
  for (int ks = 0; ks < EMB; ks += 32) {
    v16h a = frag_A(sQ, EMB, rb, ks, lane);
#pragma unroll
    for (int c = 0; c < 8; ++c) {
      v16h bmat = frag_B(sK, EMB, c * 16, ks, lane);
      acc[c] = __builtin_amdgcn_wmma_f32_16x16x32_f16(
          false, a, false, bmat, (short)0, acc[c], false, false);
    }
  }
  __syncthreads();   // all waves done reading Q/K -> safe to overlay with P

  // P = exp(S - LSEtot_row), written as f16 into this wave's 16x128 strip
  _Float16* sP = sQK + wv * 16 * BINSZ;
#pragma unroll
  for (int c = 0; c < 8; ++c) {
#pragma unroll
    for (int v = 0; v < 8; ++v) {
      float p = __expf(acc[c][v] - ltr[v]);
      sP[(v + 8 * hi) * BINSZ + c * 16 + nn] = (_Float16)p;
    }
  }

  // O = P @ V  (only this wave's strip is read -> no cross-wave barrier)
  v8f oacc[4];
#pragma unroll
  for (int c = 0; c < 4; ++c) { v8f z = {}; oacc[c] = z; }
#pragma unroll
  for (int kb = 0; kb < BINSZ; kb += 32) {
    v16h a = frag_A(sP, BINSZ, 0, kb, lane);
#pragma unroll
    for (int c = 0; c < 4; ++c) {
      v16h bmat = frag_B(sVt, BINSZ, c * 16, kb, lane);
      oacc[c] = __builtin_amdgcn_wmma_f32_16x16x32_f16(
          false, a, false, bmat, (short)0, oacc[c], false, false);
    }
  }

  // accumulate into out (tokens unique within this hash launch)
#pragma unroll
  for (int c = 0; c < 4; ++c) {
#pragma unroll
    for (int v = 0; v < 8; ++v) {
      int row = rb + v + 8 * hi;
      int tokm = tq[row];
      float* o = out + ((size_t)b * SEQ + tokm) * EMB + c * 16 + nn;
      *o += oacc[c][v];
    }
  }
}

// ---------------------------------------------------------------------------
extern "C" void kernel_launch(void* const* d_in, const int* in_sizes, int n_in,
                              void* d_out, int out_size, void* d_ws, size_t ws_size,
                              hipStream_t stream) {
  const float* q     = (const float*)d_in[0];
  const float* k     = (const float*)d_in[1];
  const float* v     = (const float*)d_in[2];
  const float* alpha = (const float*)d_in[3];
  const float* beta  = (const float*)d_in[4];
  const float* bw    = (const float*)d_in[5];
  float* out = (float*)d_out;

  char* ws = (char*)d_ws;
  size_t off = 0;
  unsigned char* bktq = (unsigned char*)(ws + off); off += (size_t)BS * NH * SEQ;        // 512K
  unsigned char* bktk = (unsigned char*)(ws + off); off += (size_t)BS * NH * SEQ;        // 512K
  int*   srcq  = (int*)(ws + off); off += (size_t)BS * HSEQ * sizeof(int);               // 2M
  int*   srck  = (int*)(ws + off); off += (size_t)BS * HSEQ * sizeof(int);               // 2M
  int*   undoq = (int*)(ws + off); off += (size_t)BS * HSEQ * sizeof(int);               // 2M
  float* lse   = (float*)(ws + off); off += (size_t)BS * HSEQ * sizeof(float);           // 2M
  float* lt    = (float*)(ws + off); off += (size_t)BS * SEQ * sizeof(float);            // 256K
  (void)ws_size; (void)in_sizes; (void)n_in; (void)out_size;

  int nout = BS * SEQ * EMB;
  k_zero<<<(nout + 255) / 256, 256, 0, stream>>>(out, nout);

  k_hash<<<BS * SEQ / TPB, 512, 0, stream>>>(q, alpha, beta, bw, bktq, 0);
  k_hash<<<BS * SEQ / TPB, 512, 0, stream>>>(k, alpha, beta, bw, bktk, 1);

  k_sort<<<BS * NH, 256, 0, stream>>>(bktq, srcq, undoq);
  k_sort<<<BS * NH, 256, 0, stream>>>(bktk, srck, nullptr);

  k_lse<<<BS * 256, 256, 0, stream>>>(q, k, srcq, srck, lse);
  k_lsetot<<<(BS * SEQ + 255) / 256, 256, 0, stream>>>(undoq, lse, lt);

  for (int h = 0; h < NH; ++h)
    k_attn<<<BS * 32, 256, 0, stream>>>(q, k, v, srcq, srck, lt, out, h);
}
